// NormalEstimation_72499047956657
// MI455X (gfx1250) — matrix operand order repaired and checked by reading, not compiled, not run on hardware
//
#include <hip/hip_runtime.h>
#include <math.h>

typedef __attribute__((ext_vector_type(16))) _Float16 v16h;
typedef __attribute__((ext_vector_type(8)))  float    v8f;

#define KNB 32
#define HID 32
#define WAVES 8          // waves per block, 1 point per wave
#define TILE_COLS 64     // per-wave staging tile: 32 rows x 64 cols (f16)

__device__ __forceinline__ void wave_sync() {
    // LDS ops from one wave are in-order in HW; this only stops the compiler
    // from reordering cross-lane LDS traffic.
    __builtin_amdgcn_fence(__ATOMIC_SEQ_CST, "wavefront");
    __builtin_amdgcn_wave_barrier();
}

// Load a 16x32 f16 A-fragment from a row-major LDS tile.
// This code's k order (same as B): lanes 0-15 hold K=0..15 (e=0..15);
// lanes 16-31 hold K=16..31. M = lane&15 (+16*rt).
__device__ __forceinline__ v16h load_a_frag(const _Float16* base, int rt, int kofs,
                                            int halfSel, int nlo) {
    v16h a;
    const _Float16* r = base + (nlo + rt * 16) * TILE_COLS + kofs + halfSel * 16;
#pragma unroll
    for (int e = 0; e < 16; ++e) {
        int kk = e;   // + halfSel*16 folded into r
        a[e] = r[kk];
    }
    return a;
}

__global__ __launch_bounds__(WAVES * 32)
void NormalEstimation_gnn_kernel(const float* __restrict__ old_weights,
                                 const float* __restrict__ pos,
                                 const float* __restrict__ normals,
                                 const int*   __restrict__ dense_l,
                                 const float* __restrict__ stddev,
                                 const float* __restrict__ W1, const float* __restrict__ b1,
                                 const float* __restrict__ W2, const float* __restrict__ b2,
                                 const float* __restrict__ W3, const float* __restrict__ b3,
                                 float* __restrict__ out_normals,
                                 float* __restrict__ out_weights,
                                 int N) {
    __shared__ __align__(16) _Float16 lds[WAVES][32 * TILE_COLS];

    const int wave = threadIdx.x >> 5;
    const int lane = threadIdx.x & 31;
    const int p    = blockIdx.x * WAVES + wave;     // point handled by this wave
    if (p >= N) return;                              // wave-uniform exit

    _Float16* tile = &lds[wave][0];
    float*    h2f  = reinterpret_cast<float*>(tile); // reused as 32x32 f32

    const int halfSel = lane >> 4;
    const int nlo     = lane & 15;

    // ---- B fragments (built once per wave). B layout: N = lane&15,
    //      lanes 0-15: K = e, lanes 16-31: K = 16+e.
    v16h bW1[2];
#pragma unroll
    for (int ct = 0; ct < 2; ++ct) {
#pragma unroll
        for (int e = 0; e < 16; ++e) {
            int kk = halfSel * 16 + e;
            int n  = nlo + 16 * ct;
            float w = (kk < 7) ? W1[kk * HID + n] : 0.0f;  // pad K 7->32
            bW1[ct][e] = (_Float16)w;
        }
    }
    v16h bW2[2][2];
#pragma unroll
    for (int kh = 0; kh < 2; ++kh)
#pragma unroll
        for (int ct = 0; ct < 2; ++ct)
#pragma unroll
            for (int e = 0; e < 16; ++e) {
                int kk = kh * 32 + halfSel * 16 + e;
                int n  = nlo + 16 * ct;
                bW2[kh][ct][e] = (_Float16)W2[kk * HID + n];
            }
    float b1v[2], b2v[2];
#pragma unroll
    for (int ct = 0; ct < 2; ++ct) {
        b1v[ct] = b1[nlo + 16 * ct];
        b2v[ct] = b2[nlo + 16 * ct];
    }
    const float w3reg = W3[lane];
    const float b3v   = b3[0];

    // ---- Phase 1: per-edge features (lane = neighbor k) ----
    const int   j   = dense_l[p * KNB + lane];
    const float pix = pos[p * 3 + 0], piy = pos[p * 3 + 1], piz = pos[p * 3 + 2];
    const float dcx = pos[j * 3 + 0] - pix;
    const float dcy = pos[j * 3 + 1] - piy;
    const float dcz = pos[j * 3 + 2] - piz;
    const float isd = 1.0f / stddev[p];
    const float dx = dcx * isd, dy = dcy * isd, dz = dcz * isd;
    const float nix = normals[p * 3 + 0], niy = normals[p * 3 + 1], niz = normals[p * 3 + 2];
    const float njx = normals[j * 3 + 0], njy = normals[j * 3 + 1], njz = normals[j * 3 + 2];
    const float f3 = old_weights[p * KNB + lane];
    const float f4 = fabsf(dx * nix + dy * niy + dz * niz);
    const float f5 = fabsf(dx * njx + dy * njy + dz * njz);
    const float f6 = fabsf(nix * njx + niy * njy + niz * njz);

    // stage features: row = lane, cols 0..6 real, 7..31 zero (K padded to 32)
    {
        _Float16* row = tile + lane * TILE_COLS;
        row[0] = (_Float16)dx; row[1] = (_Float16)dy; row[2] = (_Float16)dz;
        row[3] = (_Float16)f3; row[4] = (_Float16)f4; row[5] = (_Float16)f5;
        row[6] = (_Float16)f6;
#pragma unroll
        for (int c = 7; c < 32; ++c) row[c] = (_Float16)0.0f;
    }
    wave_sync();

    // ---- Layer 1: h = relu(f @ W1 + b1)   (4x v_wmma_f32_16x16x32_f16) ----
    v16h a1[2];
    a1[0] = load_a_frag(tile, 0, 0, halfSel, nlo);
    a1[1] = load_a_frag(tile, 1, 0, halfSel, nlo);

    v8f h1[2][2];
#pragma unroll
    for (int rt = 0; rt < 2; ++rt)
#pragma unroll
        for (int ct = 0; ct < 2; ++ct) {
            v8f z = {};
            h1[rt][ct] = __builtin_amdgcn_wmma_f32_16x16x32_f16(
                false, a1[rt], false, bW1[ct], (short)0, z, false, false);
        }
    wave_sync();

    // scatter relu(h+b1) into tile cols 0..31 (C layout: row = v + halfSel*8)
#pragma unroll
    for (int rt = 0; rt < 2; ++rt)
#pragma unroll
        for (int ct = 0; ct < 2; ++ct)
#pragma unroll
            for (int v = 0; v < 8; ++v) {
                float val = fmaxf(h1[rt][ct][v] + b1v[ct], 0.0f);
                int rowm = v + halfSel * 8 + rt * 16;
                int col  = nlo + ct * 16;
                tile[rowm * TILE_COLS + col] = (_Float16)val;
            }
    wave_sync();

    // ---- max-pool over the 32 neighbors; broadcast into cols 32..63 ----
    {
        float gmax = 0.0f;   // relu output is >= 0
#pragma unroll 8
        for (int r = 0; r < 32; ++r)
            gmax = fmaxf(gmax, (float)tile[r * TILE_COLS + lane]);
        _Float16 gh = (_Float16)gmax;
#pragma unroll 8
        for (int r = 0; r < 32; ++r)
            tile[r * TILE_COLS + 32 + lane] = gh;
    }
    wave_sync();

    // ---- Layer 2: h2 = relu([h,g] @ W2 + b2)  (8x WMMA, K=64 via 2 chained) ----
    v16h a2[2][2];
#pragma unroll
    for (int rt = 0; rt < 2; ++rt)
#pragma unroll
        for (int kh = 0; kh < 2; ++kh)
            a2[rt][kh] = load_a_frag(tile, rt, kh * 32, halfSel, nlo);
    wave_sync();   // tile is overwritten with f32 h2 below

    v8f h2[2][2];
#pragma unroll
    for (int rt = 0; rt < 2; ++rt)
#pragma unroll
        for (int ct = 0; ct < 2; ++ct) {
            v8f z = {};
            v8f t = __builtin_amdgcn_wmma_f32_16x16x32_f16(
                false, a2[rt][0], false, bW2[0][ct], (short)0, z, false, false);
            h2[rt][ct] = __builtin_amdgcn_wmma_f32_16x16x32_f16(
                false, a2[rt][1], false, bW2[1][ct], (short)0, t, false, false);
        }

#pragma unroll
    for (int rt = 0; rt < 2; ++rt)
#pragma unroll
        for (int ct = 0; ct < 2; ++ct)
#pragma unroll
            for (int v = 0; v < 8; ++v) {
                float val = fmaxf(h2[rt][ct][v] + b2v[ct], 0.0f);
                int rowm = v + halfSel * 8 + rt * 16;
                int col  = nlo + ct * 16;
                h2f[rowm * 32 + col] = val;   // f32 32x32, reuses tile storage
            }
    wave_sync();

    // ---- Layer 3: weights = sigmoid(h2 @ W3 + b3) ----
    float s = b3v;
#pragma unroll 8
    for (int cc = 0; cc < 32; ++cc) {
        int c = (cc + lane) & 31;             // rotate to avoid bank conflicts
        s += h2f[lane * 32 + c] * __shfl(w3reg, c, 32);
    }
    const float wgt = 1.0f / (1.0f + expf(-s));
    out_weights[(size_t)p * KNB + lane] = wgt;

    // ---- weighted covariance of raw offsets, wave reduction ----
    float cxx = wgt * dcx * dcx, cxy = wgt * dcx * dcy, cxz = wgt * dcx * dcz;
    float cyy = wgt * dcy * dcy, cyz = wgt * dcy * dcz, czz = wgt * dcz * dcz;
#pragma unroll
    for (int off = 16; off > 0; off >>= 1) {
        cxx += __shfl_xor(cxx, off, 32);
        cxy += __shfl_xor(cxy, off, 32);
        cxz += __shfl_xor(cxz, off, 32);
        cyy += __shfl_xor(cyy, off, 32);
        cyz += __shfl_xor(cyz, off, 32);
        czz += __shfl_xor(czz, off, 32);
    }

    if (lane == 0) {
        const float a00 = cxx + 1e-8f, a11 = cyy + 1e-8f, a22 = czz + 1e-8f;
        const float a01 = cxy, a02 = cxz, a12 = cyz;
        const float q = (a00 + a11 + a22) * (1.0f / 3.0f);
        const float t00 = a00 - q, t11 = a11 - q, t22 = a22 - q;
        const float ss = t00 * t00 + t11 * t11 + t22 * t22
                       + 2.0f * (a01 * a01 + a02 * a02 + a12 * a12);
        const float pp = sqrtf(ss * (1.0f / 6.0f)) + 1e-20f;
        const float ip = 1.0f / pp;
        const float B00 = t00 * ip, B11 = t11 * ip, B22 = t22 * ip;
        const float B01 = a01 * ip, B02 = a02 * ip, B12 = a12 * ip;
        float detB = B00 * (B11 * B22 - B12 * B12)
                   - B01 * (B01 * B22 - B12 * B02)
                   + B02 * (B01 * B12 - B11 * B02);
        float r = detB * 0.5f;
        r = fminf(fmaxf(r, -1.0f + 1e-7f), 1.0f - 1e-7f);
        const float phi = acosf(r) * (1.0f / 3.0f);
        const float e1 = q + 2.0f * pp * cosf(phi);
        const float e3 = q + 2.0f * pp * cosf(phi + 2.0943951023931953f);
        const float e2 = 3.0f * q - e1 - e3;
        float lam = e1;                                   // smallest |eig|, first wins
        if (fabsf(e2) < fabsf(lam)) lam = e2;
        if (fabsf(e3) < fabsf(lam)) lam = e3;

        const float m00 = a00 - lam, m11 = a11 - lam, m22 = a22 - lam;
        // rows: r0=(m00,a01,a02) r1=(a01,m11,a12) r2=(a02,a12,m22)
        float c0x = a01 * a12 - a02 * m11, c0y = a02 * a01 - m00 * a12, c0z = m00 * m11 - a01 * a01;
        float c1x = m11 * m22 - a12 * a12, c1y = a12 * a02 - a01 * m22, c1z = a01 * a12 - m11 * a02;
        float c2x = a12 * a01 - m22 * m00, c2y = a02 * a02 - m22 * m22 + m22 * m22 - m00 * m22, c2z = a02 * a12 - a12 * a02;
        // recompute c2 = r2 x r0 carefully:
        c2x = a12 * a02 - m22 * a01;
        c2y = m22 * m00 - a02 * a02;
        c2z = a02 * a01 - a12 * m00;
        float n0 = sqrtf(c0x * c0x + c0y * c0y + c0z * c0z);
        float n1 = sqrtf(c1x * c1x + c1y * c1y + c1z * c1z);
        float n2 = sqrtf(c2x * c2x + c2y * c2y + c2z * c2z);
        float vx = c0x, vy = c0y, vz = c0z, bn = n0;
        if (n1 > bn) { vx = c1x; vy = c1y; vz = c1z; bn = n1; }
        if (n2 > bn) { vx = c2x; vy = c2y; vz = c2z; bn = n2; }
        const float inorm = 1.0f / (sqrtf(vx * vx + vy * vy + vz * vz) + 1e-12f);
        out_normals[(size_t)p * 3 + 0] = vx * inorm;
        out_normals[(size_t)p * 3 + 1] = vy * inorm;
        out_normals[(size_t)p * 3 + 2] = vz * inorm;
    }
}

extern "C" void kernel_launch(void* const* d_in, const int* in_sizes, int n_in,
                              void* d_out, int out_size, void* d_ws, size_t ws_size,
                              hipStream_t stream) {
    const float* old_weights = (const float*)d_in[0];
    const float* pos         = (const float*)d_in[1];
    // d_in[2] = batch (unused)
    const float* normals     = (const float*)d_in[3];
    // d_in[4] = edge_idx_l (unused; dense_l carries the same info)
    const int*   dense_l     = (const int*)d_in[5];
    const float* stddev      = (const float*)d_in[6];
    const float* W1 = (const float*)d_in[7];
    const float* b1 = (const float*)d_in[8];
    const float* W2 = (const float*)d_in[9];
    const float* b2 = (const float*)d_in[10];
    const float* W3 = (const float*)d_in[11];
    const float* b3 = (const float*)d_in[12];

    const int N = in_sizes[1] / 3;                 // pos is (N,3)
    float* out_normals = (float*)d_out;            // first N*3 floats
    float* out_weights = out_normals + (size_t)N * 3;  // then N*K floats

    const int blocks = (N + WAVES - 1) / WAVES;    // 1 point per wave, 8 waves/block
    NormalEstimation_gnn_kernel<<<blocks, WAVES * 32, 0, stream>>>(
        old_weights, pos, normals, dense_l, stddev,
        W1, b1, W2, b2, W3, b3, out_normals, out_weights, N);
}
